// SelectiveSSM_44736379355743
// MI455X (gfx1250) — matrix operand order repaired and checked by reading, not compiled, not run on hardware
//
#include <hip/hip_runtime.h>
#include <math.h>

// ---------------------------------------------------------------------------
// Problem constants (B=1)
#define LL  2048   // sequence length
#define DM  1024   // model dim
#define DI  2048   // inner dim
#define NS  16     // SSM state size
#define KC  4      // conv kernel
#define XR_W (2*DI)       // 4096
#define PJ_W (DI + 2*NS)  // 2080
#define CH  64            // scan chunks
#define CT  (LL/CH)       // 32 steps per chunk

typedef __bf16 bf16;
typedef __attribute__((ext_vector_type(16))) __bf16 v16bf;
typedef __attribute__((ext_vector_type(8)))  __bf16 v8bf;
typedef __attribute__((ext_vector_type(8)))  float  v8f;

union AFrag { v16bf v; v8bf h[2]; };

// ---------------------------------------------------------------------------
// C[M,N] = A[M,K](bf16) * W[N,K](bf16)^T + bias[N], f32 out.
// One wave computes a (16*FM) x 32 tile: 2*FM WMMAs per (FM+2) fragment loads.
// launch_bounds(...,1) relaxes the occupancy-driven VGPR cap so all fragments
// of a k-step stay live in disjoint registers (loads batch into one clause,
// WMMAs issue back-to-back with staggered loadcnt waits, no WAR v_nops).
// M multiple of 16*FM, N and K multiples of 32.
template<int FM>
__global__ void __launch_bounds__(256, 1)
gemm_bf16_wmma(const bf16* __restrict__ A,
               const bf16* __restrict__ W,
               const float* __restrict__ bias,
               float* __restrict__ C,
               int M, int N, int K) {
  const int lane = threadIdx.x & 31;
  const int wave = threadIdx.x >> 5;
  const int tilesN = N >> 5;
  const int totalTiles = (M / (16 * FM)) * tilesN;
  const int tile = blockIdx.x * (blockDim.x >> 5) + wave;
  if (tile >= totalTiles) return;          // wave-uniform: EXEC stays all-1s

  const int m0 = (tile / tilesN) * (16 * FM);
  const int n0 = (tile % tilesN) << 5;

  // Per-lane base pointers (this code's own contiguous k order k = 16h + s, same for A and B).
  // A (16x32, MxK): lanes 0-15 row=lane, K=0..15; lanes 16-31 same
  //                 rows, K=16..31  -> two 16B loads at +0 / +8 elems.
  // B (32x16, KxN): lanes 0-15 col=lane, K=0..15; lanes 16-31 K=16..31
  //                 -> one 32B load.
  const int rsel  = lane & 15;
  const int hi    = lane >> 4;
  const bf16* ap[FM];
#pragma unroll
  for (int i = 0; i < FM; ++i)
    ap[i] = A + (size_t)(m0 + 16 * i + rsel) * K + (hi ? 16 : 0);
  const bf16* bp0 = W + (size_t)(n0 + rsel) * K + (hi ? 16 : 0);
  const bf16* bp1 = W + (size_t)(n0 + 16 + rsel) * K + (hi ? 16 : 0);

  v8f acc[FM][2];
#pragma unroll
  for (int i = 0; i < FM; ++i) { acc[i][0] = (v8f){}; acc[i][1] = (v8f){}; }

#pragma unroll 2
  for (int k0 = 0; k0 < K; k0 += 32) {
    // Load phase: all fragments for this k-step.
    v16bf b0 = *(const v16bf*)(bp0 + k0);
    v16bf b1 = *(const v16bf*)(bp1 + k0);
    v16bf a[FM];
#pragma unroll
    for (int i = 0; i < FM; ++i) {
      AFrag f;
      f.h[0] = *(const v8bf*)(ap[i] + k0);
      f.h[1] = *(const v8bf*)(ap[i] + k0 + 8);
      a[i] = f.v;
    }
    // Compute phase: 2*FM back-to-back WMMAs.
#pragma unroll
    for (int i = 0; i < FM; ++i) {
      acc[i][0] = __builtin_amdgcn_wmma_f32_16x16x32_bf16(false, a[i], false, b0,
                                      (short)0, acc[i][0], false, false);
      acc[i][1] = __builtin_amdgcn_wmma_f32_16x16x32_bf16(false, a[i], false, b1,
                                      (short)0, acc[i][1], false, false);
    }
  }

  // C/D layout: VGPR j -> (row = j + (lane>=16 ? 8 : 0), col = lane&15)
  const int rowOff = hi << 3;
  const float bLo = bias[n0 + rsel];
  const float bHi = bias[n0 + 16 + rsel];
#pragma unroll
  for (int i = 0; i < FM; ++i) {
#pragma unroll
    for (int j = 0; j < 8; ++j) {
      int r = m0 + 16 * i + rowOff + j;
      C[(size_t)r * N + n0 + rsel]      = acc[i][0][j] + bLo;
      C[(size_t)r * N + n0 + 16 + rsel] = acc[i][1][j] + bHi;
    }
  }
}

// ---------------------------------------------------------------------------
__global__ void f32_to_bf16(const float* __restrict__ in, bf16* __restrict__ out,
                            int n) {
  int i = blockIdx.x * blockDim.x + threadIdx.x;
  if (i < n) out[i] = (bf16)in[i];
}

// Depthwise causal conv (K=4) over L + bias + SiLU.
// xr layout: (L, 4096); channels 0..2047 are the conv input, 2048.. is `res`.
__global__ void conv_silu(const float* __restrict__ xr,
                          const float* __restrict__ Wc,
                          const float* __restrict__ bc,
                          float* __restrict__ xi,
                          bf16* __restrict__ xi_bf) {
  int idx = blockIdx.x * blockDim.x + threadIdx.x;   // over L*DI
  if (idx >= LL * DI) return;
  int d = idx & (DI - 1);
  int l = idx >> 11;                                 // DI = 2^11
  float acc = bc[d];
#pragma unroll
  for (int k = 0; k < KC; ++k) {
    int ls = l - (KC - 1) + k;
    if (ls >= 0) acc = fmaf(Wc[d * KC + k], xr[(size_t)ls * XR_W + d], acc);
  }
  float s = acc / (1.f + __expf(-acc));              // SiLU
  xi[idx]    = s;
  xi_bf[idx] = (bf16)s;
}

// ---------------------------------------------------------------------------
// Chunked parallel scan over the linear recurrence h[l] = a[l]*h[l-1] + b[l].
// Phase 1: per chunk c (length CT), run with h0=0; emit local end-state and
//          the cumulative coefficient product P = prod_l a[l].
__global__ void scan_part1(const float* __restrict__ proj,
                           const float* __restrict__ xi,
                           const float* __restrict__ A_log,
                           float* __restrict__ hloc,
                           float* __restrict__ Pa) {
  const int d = blockIdx.x * blockDim.x + threadIdx.x;   // channel
  const int c = blockIdx.y;                               // chunk
  __shared__ float sB[NS];

  float Aneg[NS], h[NS], P[NS];
#pragma unroll
  for (int n = 0; n < NS; ++n) {
    Aneg[n] = -__expf(A_log[d * NS + n]);
    h[n] = 0.f; P[n] = 1.f;
  }

  for (int t = 0; t < CT; ++t) {
    const int l = c * CT + t;
    if (threadIdx.x < NS)
      sB[threadIdx.x] = proj[(size_t)l * PJ_W + DI + threadIdx.x];
    __syncthreads();

    float p = proj[(size_t)l * PJ_W + d];
    float delta = (p > 20.f) ? p : log1pf(__expf(p));    // softplus
    float dbx = delta * xi[(size_t)l * DI + d];
#pragma unroll
    for (int n = 0; n < NS; ++n) {
      float da = __expf(delta * Aneg[n]);
      h[n] = fmaf(da, h[n], dbx * sB[n]);
      P[n] *= da;
    }
    __syncthreads();
  }

  const size_t base = ((size_t)c * DI + d) * NS;
#pragma unroll
  for (int n = 0; n < NS; ++n) { hloc[base + n] = h[n]; Pa[base + n] = P[n]; }
}

// Phase 2: scan over the CH chunk summaries, per (d,n) independently.
// H0[c] = initial state of chunk c;  H0[c+1] = Pa[c]*H0[c] + hloc[c].
__global__ void scan_part2(const float* __restrict__ hloc,
                           const float* __restrict__ Pa,
                           float* __restrict__ H0) {
  int i = blockIdx.x * blockDim.x + threadIdx.x;         // over DI*NS
  if (i >= DI * NS) return;
  float run = 0.f;
  for (int c = 0; c < CH; ++c) {
    size_t idx = (size_t)c * DI * NS + i;
    H0[idx] = run;
    run = fmaf(Pa[idx], run, hloc[idx]);
  }
}

// Phase 3: replay each chunk from its true initial state; emit
// z = (y + xi*D) * silu(res) as bf16 for the final GEMM.
__global__ void scan_part3(const float* __restrict__ proj,
                           const float* __restrict__ xi,
                           const float* __restrict__ xr,
                           const float* __restrict__ A_log,
                           const float* __restrict__ Dp,
                           const float* __restrict__ H0,
                           bf16* __restrict__ z_bf) {
  const int d = blockIdx.x * blockDim.x + threadIdx.x;
  const int c = blockIdx.y;
  __shared__ float sBC[2 * NS];

  float Aneg[NS], h[NS];
  const size_t base = ((size_t)c * DI + d) * NS;
#pragma unroll
  for (int n = 0; n < NS; ++n) {
    Aneg[n] = -__expf(A_log[d * NS + n]);
    h[n] = H0[base + n];
  }
  const float Dd = Dp[d];

  for (int t = 0; t < CT; ++t) {
    const int l = c * CT + t;
    if (threadIdx.x < 2 * NS)
      sBC[threadIdx.x] = proj[(size_t)l * PJ_W + DI + threadIdx.x];
    __syncthreads();

    float p = proj[(size_t)l * PJ_W + d];
    float delta = (p > 20.f) ? p : log1pf(__expf(p));
    float xv = xi[(size_t)l * DI + d];
    float dbx = delta * xv;
    float y = 0.f;
#pragma unroll
    for (int n = 0; n < NS; ++n) {
      float da = __expf(delta * Aneg[n]);
      h[n] = fmaf(da, h[n], dbx * sBC[n]);
      y = fmaf(h[n], sBC[NS + n], y);
    }
    float res = xr[(size_t)l * XR_W + DI + d];
    float z = (y + xv * Dd) * (res / (1.f + __expf(-res)));
    z_bf[(size_t)l * DI + d] = (bf16)z;
    __syncthreads();
  }
}

// ---------------------------------------------------------------------------
extern "C" void kernel_launch(void* const* d_in, const int* in_sizes, int n_in,
                              void* d_out, int out_size, void* d_ws, size_t ws_size,
                              hipStream_t stream) {
  const float* x      = (const float*)d_in[0];
  const float* W_in   = (const float*)d_in[1];
  const float* b_in   = (const float*)d_in[2];
  const float* W_conv = (const float*)d_in[3];
  const float* b_conv = (const float*)d_in[4];
  const float* W_x    = (const float*)d_in[5];
  const float* b_x    = (const float*)d_in[6];
  const float* W_out  = (const float*)d_in[7];
  const float* b_out  = (const float*)d_in[8];
  const float* A_log  = (const float*)d_in[9];
  const float* Dp     = (const float*)d_in[10];
  float* out = (float*)d_out;

  // Workspace carve-up (all chunks multiples of 256B; base is hipMalloc-aligned)
  char* p = (char*)d_ws;
  bf16*  x_bf    = (bf16*)p;  p += (size_t)LL * DM * 2;        //  4   MB
  bf16*  Win_bf  = (bf16*)p;  p += (size_t)XR_W * DM * 2;      //  8   MB
  bf16*  Wx_bf   = (bf16*)p;  p += (size_t)PJ_W * DI * 2;      //  8.5 MB
  bf16*  Wout_bf = (bf16*)p;  p += (size_t)DM * DI * 2;        //  4   MB
  float* xr      = (float*)p; p += (size_t)LL * XR_W * 4;      // 32   MB
  float* xi      = (float*)p; p += (size_t)LL * DI * 4;        // 16   MB
  bf16*  xi_bf   = (bf16*)p;  p += (size_t)LL * DI * 2;        //  8   MB
  float* proj    = (float*)p; p += (size_t)LL * PJ_W * 4;      // 17   MB
  bf16*  z_bf    = (bf16*)p;  p += (size_t)LL * DI * 2;        //  8   MB
  float* hloc    = (float*)p; p += (size_t)CH * DI * NS * 4;   //  8   MB
  float* Pa      = (float*)p; p += (size_t)CH * DI * NS * 4;   //  8   MB
  float* H0      = (float*)p; p += (size_t)CH * DI * NS * 4;   //  8   MB

  const int CVT_B = 256;
  // fp32 -> bf16 for GEMM operands
  {
    int n;
    n = LL * DM;     f32_to_bf16<<<(n + CVT_B - 1) / CVT_B, CVT_B, 0, stream>>>(x,     x_bf,    n);
    n = XR_W * DM;   f32_to_bf16<<<(n + CVT_B - 1) / CVT_B, CVT_B, 0, stream>>>(W_in,  Win_bf,  n);
    n = PJ_W * DI;   f32_to_bf16<<<(n + CVT_B - 1) / CVT_B, CVT_B, 0, stream>>>(W_x,   Wx_bf,   n);
    n = DM * DI;     f32_to_bf16<<<(n + CVT_B - 1) / CVT_B, CVT_B, 0, stream>>>(W_out, Wout_bf, n);
  }

  // GEMM1: xr = x @ W_in^T + b_in     (M=2048, N=4096, K=1024), 64x32 tiles
  {
    int tiles = (LL / 64) * (XR_W >> 5);
    gemm_bf16_wmma<4><<<(tiles + 7) / 8, 256, 0, stream>>>(x_bf, Win_bf, b_in, xr,
                                                           LL, XR_W, DM);
  }

  // Depthwise conv + SiLU -> xi (f32 + bf16)
  {
    int n = LL * DI;
    conv_silu<<<(n + 255) / 256, 256, 0, stream>>>(xr, W_conv, b_conv, xi, xi_bf);
  }

  // GEMM2: proj = xi @ W_x^T + b_x    (M=2048, N=2080, K=2048)
  {
    int tiles = (LL / 64) * (PJ_W >> 5);
    gemm_bf16_wmma<4><<<(tiles + 7) / 8, 256, 0, stream>>>(xi_bf, Wx_bf, b_x, proj,
                                                           LL, PJ_W, DI);
  }

  // Chunked parallel scan: 64 chunks x 32 steps, 131k threads in phases 1/3.
  {
    dim3 grid(DI / 256, CH);
    scan_part1<<<grid, 256, 0, stream>>>(proj, xi, A_log, hloc, Pa);
    scan_part2<<<(DI * NS + 255) / 256, 256, 0, stream>>>(hloc, Pa, H0);
    scan_part3<<<grid, 256, 0, stream>>>(proj, xi, xr, A_log, Dp, H0, z_bf);
  }

  // GEMM3: out = z @ W_out^T + b_out  (M=2048, N=1024, K=2048)
  {
    int tiles = (LL / 64) * (DM >> 5);
    gemm_bf16_wmma<4><<<(tiles + 7) / 8, 256, 0, stream>>>(z_bf, Wout_bf, b_out, out,
                                                           LL, DM, DI);
  }
}
